// MultiHeadAttention_9105330667527
// MI455X (gfx1250) — matrix operand order repaired and checked by reading, not compiled, not run on hardware
//
#include <hip/hip_runtime.h>

// ---------------------------------------------------------------------------
// MHA forward for MI455X (gfx1250, wave32, WMMA 16x16x32 f16 -> f32)
//   B=4, S=2048, HIDDEN=768, HEADS=12, DHEAD=64
// ---------------------------------------------------------------------------

typedef __attribute__((ext_vector_type(16))) _Float16 v16h;
typedef __attribute__((ext_vector_type(8)))  _Float16 v8h;
typedef __attribute__((ext_vector_type(8)))  float    v8f;
typedef __attribute__((ext_vector_type(4)))  int      v4i;

#define HID  768
#define NH   12
#define DH   64
#define SLEN 2048
#define BATCH 4

union V16 { v16h v; v8h h[2]; };

static __device__ __forceinline__ v8f wmma_f16(v16h a, v16h b, v8f c) {
  return __builtin_amdgcn_wmma_f32_16x16x32_f16(false, a, false, b, (short)0, c,
                                                false, false);
}

// ---- CDNA5 async global->LDS copy (ASYNCcnt-tracked), 16B per lane --------
typedef __attribute__((address_space(1))) v4i* gvec_p;
typedef __attribute__((address_space(3))) v4i* lvec_p;

static __device__ __forceinline__ void async_ld16(_Float16* lds,
                                                  const _Float16* g) {
#if __has_builtin(__builtin_amdgcn_global_load_async_to_lds_b128)
  __builtin_amdgcn_global_load_async_to_lds_b128(
      (gvec_p)(_Float16*)g, (lvec_p)lds, 0, 0);
#else
  asm volatile("global_load_async_to_lds_b128 %0, %1, off"
               :: "v"((unsigned)(unsigned long)(lvec_p)lds),
                  "v"(g)
               : "memory");
#endif
}

static __device__ __forceinline__ void wait_async2() {
#if __has_builtin(__builtin_amdgcn_s_wait_asynccnt)
  __builtin_amdgcn_s_wait_asynccnt(2);
#else
  asm volatile("s_wait_asynccnt 0x2" ::: "memory");
#endif
}
static __device__ __forceinline__ void wait_async0() {
#if __has_builtin(__builtin_amdgcn_s_wait_asynccnt)
  __builtin_amdgcn_s_wait_asynccnt(0);
#else
  asm volatile("s_wait_asynccnt 0x0" ::: "memory");
#endif
}

// ---------------------------------------------------------------------------
// Projection GEMM: C[m,n] = sum_k X[m,k] * W[n,k]  (+bias, +u, *1/8 for Q)
// One wave per 16x16 tile, 24 WMMA k-steps. Q -> qh f16 [B,H,S,D];
// V -> vt f16 [B,H,D,S] (pre-transposed for the PV WMMA B-operand).
// ---------------------------------------------------------------------------
__global__ __launch_bounds__(256) void proj_wmma_kernel(
    const float* __restrict__ query, const float* __restrict__ value,
    const float* __restrict__ Wq, const float* __restrict__ Wv,
    const float* __restrict__ bq, const float* __restrict__ bv,
    const float* __restrict__ u,
    _Float16* __restrict__ qh, _Float16* __restrict__ vt)
{
  const bool isV = (blockIdx.y != 0);
  const float* __restrict__ X    = isV ? value : query;
  const float* __restrict__ W    = isV ? Wv : Wq;
  const float* __restrict__ bias = isV ? bv : bq;

  const int wave = threadIdx.x >> 5;
  const int lane = threadIdx.x & 31;
  const int rr   = lane & 15;
  const bool hi  = lane >= 16;

  const int tile = blockIdx.x * 8 + wave;      // 512*48 tiles total
  const int nt = tile % (HID / 16);
  const int mt = tile / (HID / 16);
  const int m0 = mt * 16, n0 = nt * 16;

  const float* __restrict__ xrow = X + (size_t)(m0 + rr) * HID;
  const float* __restrict__ wrow = W + (size_t)(n0 + rr) * HID;
  const int akoff = hi ? 16 : 0;
  const int bkoff = hi ? 16 : 0;

  v8f c = {};
  for (int kb = 0; kb < HID; kb += 32) {
    V16 A, Bm;
#pragma unroll
    for (int i = 0; i < 8; ++i)  A.v[i]     = (_Float16)xrow[kb + akoff + i];
#pragma unroll
    for (int i = 0; i < 8; ++i)  A.v[8 + i] = (_Float16)xrow[kb + 8 + akoff + i];
#pragma unroll
    for (int i = 0; i < 16; ++i) Bm.v[i]    = (_Float16)wrow[kb + bkoff + i];
    c = wmma_f16(A.v, Bm.v, c);
  }

  const int n = n0 + rr;
  const int h = n / DH, d = n % DH;
  const int b = m0 / SLEN;
  const int srow0 = (m0 % SLEN) + (hi ? 8 : 0);
  const size_t bh = (size_t)b * NH + h;

  if (!isV) {
    const float add = bias[n] + u[n];
    _Float16* dst = qh + (bh * SLEN + srow0) * DH + d;
#pragma unroll
    for (int i = 0; i < 8; ++i)
      dst[(size_t)i * DH] = (_Float16)((c[i] + add) * 0.125f);
  } else {
    const float add = bias[n];
    _Float16* dst = vt + (bh * DH + d) * SLEN + srow0;
#pragma unroll
    for (int i = 0; i < 8; ++i)
      dst[i] = (_Float16)(c[i] + add);
  }
}

// ---------------------------------------------------------------------------
// Key: fp32 [B,S,HID] -> f16 [B,H,S,D]
// ---------------------------------------------------------------------------
__global__ __launch_bounds__(256) void k_convert_kernel(
    const float* __restrict__ key, _Float16* __restrict__ kh)
{
  const int idx = blockIdx.x * 256 + threadIdx.x;
  const int d = idx & (DH - 1);
  const int s = (idx / DH) & (SLEN - 1);
  const int h = (idx / (DH * SLEN)) % NH;
  const int b = idx / (DH * SLEN * NH);
  kh[idx] = (_Float16)key[((size_t)b * SLEN + s) * HID + h * DH + d];
}

// ---------------------------------------------------------------------------
// Flash attention: 8 waves/block share one (b,h); per-block async double-
// buffered LDS staging of 32-key K/V blocks; per-wave 16-row Q tile.
// ---------------------------------------------------------------------------
#define KSTR 72   // LDS row stride (halves) for K stage: 16B aligned, no conflicts
#define VSTR 40   // LDS row stride (halves) for V stage

__global__ __launch_bounds__(256) void flash_attn_kernel(
    const _Float16* __restrict__ qh, const _Float16* __restrict__ kh,
    const _Float16* __restrict__ vt, float* __restrict__ out)
{
  __shared__ _Float16 Ks[2][32 * KSTR];   // staged K block  [32 keys][64 d]
  __shared__ _Float16 Vs[2][64 * VSTR];   // staged Vt block [64 d][32 keys]
  __shared__ _Float16 Pl[8][16 * 40];     // per-wave P transpose tile

  const int tid  = threadIdx.x;
  const int wave = tid >> 5;
  const int lane = tid & 31;
  const int rr   = lane & 15;
  const bool hi  = lane >= 16;

  const int qt = blockIdx.x * 8 + wave;       // 6144 q-tiles total
  const int bh = qt >> 7;                     // same for all 8 waves in block
  const int q0 = (qt & 127) * 16;
  const int b = bh / NH, h = bh % NH;

  const _Float16* __restrict__ kbase = kh + (size_t)bh * SLEN * DH;
  const _Float16* __restrict__ vbase = vt + (size_t)bh * DH * SLEN;

  // cooperative-copy coordinates (16B per thread per matrix)
  const int krow = tid >> 3, kchunk = (tid & 7) * 8;   // 32 rows x 64 halves
  const int vrow = tid >> 2, vchunk = (tid & 3) * 8;   // 64 rows x 32 halves

  // Q tile as two A operands (d 0..31 and 32..63), resident for whole loop
  const _Float16* __restrict__ qrow = qh + ((size_t)bh * SLEN + q0 + rr) * DH;
  const int akoff = hi ? 16 : 0;
  V16 Aq0, Aq1;
  Aq0.h[0] = *(const v8h*)(qrow + akoff);
  Aq0.h[1] = *(const v8h*)(qrow + 8 + akoff);
  Aq1.h[0] = *(const v8h*)(qrow + 32 + akoff);
  Aq1.h[1] = *(const v8h*)(qrow + 40 + akoff);

  float m_i[8], l_i[8];
  v8f acc[4] = {};
#pragma unroll
  for (int i = 0; i < 8; ++i) { m_i[i] = -3.0e38f; l_i[i] = 0.0f; }

  _Float16* Plw = &Pl[wave][0];
  const int koffB = hi ? 16 : 0;

  // prologue: async-stage first K/V block into buffer 0
  async_ld16(&Ks[0][krow * KSTR + kchunk],
             kbase + (size_t)krow * DH + kchunk);
  async_ld16(&Vs[0][vrow * VSTR + vchunk],
             vbase + (size_t)vrow * SLEN + vchunk);

  int cur = 0;
  for (int kb = 0; kb < SLEN; kb += 32) {
    // issue next block's DMA into the other buffer, then wait for current
    if (kb + 32 < SLEN) {
      const int nxt = cur ^ 1;
      async_ld16(&Ks[nxt][krow * KSTR + kchunk],
                 kbase + (size_t)(kb + 32 + krow) * DH + kchunk);
      async_ld16(&Vs[nxt][vrow * VSTR + vchunk],
                 vbase + (size_t)vrow * SLEN + (kb + 32) + vchunk);
      wait_async2();                 // current block's 2 copies are done
    } else {
      wait_async0();
    }
    __syncthreads();                 // cross-wave visibility of staged data

    // ---- scores from LDS: two 16(q) x 16(key) tiles over 32 keys ----
    const _Float16* kr0 = &Ks[cur][(size_t)rr * KSTR];
    const _Float16* kr1 = &Ks[cur][(size_t)(16 + rr) * KSTR];
    const v16h Bk0a = *(const v16h*)(kr0 + koffB);
    const v16h Bk0b = *(const v16h*)(kr0 + 32 + koffB);
    const v16h Bk1a = *(const v16h*)(kr1 + koffB);
    const v16h Bk1b = *(const v16h*)(kr1 + 32 + koffB);

    v8f z = {};
    v8f s0 = wmma_f16(Aq1.v, Bk0b, wmma_f16(Aq0.v, Bk0a, z));
    v8f s1 = wmma_f16(Aq1.v, Bk1b, wmma_f16(Aq0.v, Bk1a, z));

    // ---- online softmax; each C row lives in 16 lanes of one half-wave ----
#pragma unroll
    for (int i = 0; i < 8; ++i) {
      const float v0 = s0[i], v1 = s1[i];
      float mx = fmaxf(v0, v1);
      mx = fmaxf(mx, __shfl_xor(mx, 1));
      mx = fmaxf(mx, __shfl_xor(mx, 2));
      mx = fmaxf(mx, __shfl_xor(mx, 4));
      mx = fmaxf(mx, __shfl_xor(mx, 8));
      const float mnew  = fmaxf(m_i[i], mx);
      const float alpha = __expf(m_i[i] - mnew);
      const float p0 = __expf(v0 - mnew);
      const float p1 = __expf(v1 - mnew);
      float rs = p0 + p1;
      rs += __shfl_xor(rs, 1);
      rs += __shfl_xor(rs, 2);
      rs += __shfl_xor(rs, 4);
      rs += __shfl_xor(rs, 8);
      l_i[i] = l_i[i] * alpha + rs;
      m_i[i] = mnew;
      acc[0][i] *= alpha; acc[1][i] *= alpha;
      acc[2][i] *= alpha; acc[3][i] *= alpha;

      const int row = i + (hi ? 8 : 0);
      Plw[row * 40 + rr]      = (_Float16)p0;
      Plw[row * 40 + 16 + rr] = (_Float16)p1;
    }

    // ---- reload P in the 16x32 A-operand layout (in-wave, DS-ordered) ----
    const v8h* prow = (const v8h*)(Plw + rr * 40);
    V16 Ap;
    Ap.h[0] = prow[hi ? 2 : 0];
    Ap.h[1] = prow[hi ? 3 : 1];

    // ---- PV from LDS: acc[c] += P(16x32) @ V(32 x 16d) ----
#pragma unroll
    for (int c = 0; c < 4; ++c) {
      const v16h Bv = *(const v16h*)(&Vs[cur][(size_t)(c * 16 + rr) * VSTR] +
                                     koffB);
      acc[c] = wmma_f16(Ap.v, Bv, acc[c]);
    }

    __syncthreads();                 // all reads done before buffer reuse
    cur ^= 1;
  }

  // ---- normalize and write fp32 [B,S,HID] ----
  float* orow = out + ((size_t)b * SLEN + q0 + (hi ? 8 : 0)) * HID + h * DH + rr;
#pragma unroll
  for (int i = 0; i < 8; ++i) {
    const float inv = 1.0f / l_i[i];
#pragma unroll
    for (int c = 0; c < 4; ++c)
      orow[(size_t)i * HID + c * 16] = acc[c][i] * inv;
  }
}

// ---------------------------------------------------------------------------
extern "C" void kernel_launch(void* const* d_in, const int* in_sizes, int n_in,
                              void* d_out, int out_size, void* d_ws, size_t ws_size,
                              hipStream_t stream) {
  (void)in_sizes; (void)n_in; (void)out_size; (void)ws_size;
  const float* query = (const float*)d_in[0];
  const float* key   = (const float*)d_in[1];
  const float* value = (const float*)d_in[2];
  const float* Wq    = (const float*)d_in[3];
  const float* bq    = (const float*)d_in[4];
  const float* Wv    = (const float*)d_in[5];
  const float* bv    = (const float*)d_in[6];
  const float* u     = (const float*)d_in[7];
  float* out = (float*)d_out;

  const size_t elems = (size_t)BATCH * SLEN * HID;   // 6,291,456
  char* ws = (char*)d_ws;
  _Float16* qh = (_Float16*)ws;                              // [B,H,S,D] f16
  _Float16* kh = (_Float16*)(ws + elems * sizeof(_Float16)); // [B,H,S,D] f16
  _Float16* vt = (_Float16*)(ws + 2 * elems * sizeof(_Float16)); // [B,H,D,S]

  dim3 gproj((BATCH * SLEN / 16) * (HID / 16) / 8, 2);   // 3072 x 2 blocks
  proj_wmma_kernel<<<gproj, 256, 0, stream>>>(query, value, Wq, Wv, bq, bv, u,
                                              qh, vt);
  k_convert_kernel<<<(int)(elems / 256), 256, 0, stream>>>(key, kh);
  flash_attn_kernel<<<(BATCH * NH * (SLEN / 16)) / 8, 256, 0, stream>>>(qh, kh,
                                                                        vt, out);
}
